// DeformableDetrModel_17514876634186
// MI455X (gfx1250) — matrix operand order repaired and checked by reading, not compiled, not run on hardware
//
#include <hip/hip_runtime.h>

#define D_MODEL 256
#define N_HEADS 8
#define N_POINTS 4
#define D_HEAD 32
#define NQ 300
#define BS 16

typedef __attribute__((ext_vector_type(16))) _Float16 v16h;
typedef __attribute__((ext_vector_type(8)))  float    v8f;

// ---- workspace layout (bytes) ----
// [0, 128K)           : G, f32 [16][8][256]
// [128K, 144K)        : Fa  W_attn fragments, 16 frags  (kc*2 + t)
// [144K, 272K)        : Fv  W_val  fragments, 128 frags (kc*16 + t)
// [272K, 400K)        : Fo  W_out  fragments, 128 frags (h*16 + t)
#define WS_G_BYTES   (16 * N_HEADS * D_MODEL * 4)
#define WS_FA_OFF    WS_G_BYTES
#define WS_FV_OFF    (WS_FA_OFF + 16 * 32 * 32)
#define WS_FO_OFF    (WS_FV_OFF + 128 * 32 * 32)

#define NFRAG_A 16
#define NFRAG_V 128
#define NFRAG_O 128
#define NFRAG_TOTAL (NFRAG_A + NFRAG_V + NFRAG_O)

__device__ __forceinline__ v8f wmma_f32_f16(v16h a, v16h b, v8f c) {
  return __builtin_amdgcn_wmma_f32_16x16x32_f16(
      /*neg_a=*/false, a, /*neg_b=*/false, b,
      /*c_mod=*/(short)0, c, /*reuse_a=*/false, /*reuse_b=*/false);
}

// A-fragment: 16x32 tile, f16, from row-major f32 (row stride ld), K offset k0.
// lane 0-15 -> M=lane, K in {0..15}; lane 16-31 -> K in {16..31} (matches B swizzle).
__device__ __forceinline__ v16h load_a_f32(const float* A, int ld, int k0, int lane) {
  const int M  = lane & 15;
  const int kb = (lane & 16);             // 0 or 16
  const float* r = A + M * ld + k0 + kb;
  v16h a;
#pragma unroll
  for (int e = 0; e < 8; ++e) a[e]     = (_Float16)r[e];        // K = kb+0..7
#pragma unroll
  for (int e = 0; e < 8; ++e) a[8 + e] = (_Float16)r[8 + e];    // K = kb+8..15
  return a;
}

// ---------------------------------------------------------------------------
// Kernel 0: swizzle weights (row-major KxN f32) into per-lane WMMA B-fragment
// layout, f16, lane-major (1 KB per fragment, fully coalesced on read).
// B layout: N = lane&15; K = k0 + (lane&16) + e.
// ---------------------------------------------------------------------------
__global__ __launch_bounds__(256) void k_swizzle_weights(
    const float* __restrict__ Wa, const float* __restrict__ Wv,
    const float* __restrict__ Wo, v16h* __restrict__ Fa,
    v16h* __restrict__ Fv, v16h* __restrict__ Fo) {
  const int g = blockIdx.x * blockDim.x + threadIdx.x;   // one (frag,lane) each
  if (g >= NFRAG_TOTAL * 32) return;
  const int lane = g & 31;
  const int f    = g >> 5;

  const float* W;
  v16h* F;
  int fi, k0, t, ldn;
  if (f < NFRAG_A) {                       // W_attn: frag = kc*2 + t
    fi = f;  W = Wa;  F = Fa;  ldn = N_HEADS * N_POINTS;
    k0 = (fi >> 1) * 32;  t = fi & 1;
  } else if (f < NFRAG_A + NFRAG_V) {      // W_val: frag = kc*16 + t
    fi = f - NFRAG_A;  W = Wv;  F = Fv;  ldn = D_MODEL;
    k0 = (fi >> 4) * 32;  t = fi & 15;
  } else {                                 // W_out: frag = h*16 + t, K block = h*32
    fi = f - NFRAG_A - NFRAG_V;  W = Wo;  F = Fo;  ldn = D_MODEL;
    k0 = (fi >> 4) * D_HEAD;  t = fi & 15;
  }

  const int N  = t * 16 + (lane & 15);
  const int kr = k0 + (lane & 16);
  v16h b;
#pragma unroll
  for (int e = 0; e < 16; ++e) b[e] = (_Float16)W[(kr + e) * ldn + N];
  F[fi * 32 + lane] = b;                   // 32B/lane, lane-major -> coalesced
}

// ---------------------------------------------------------------------------
// Kernel 1 (1 block): mem_feat = memory[0] @ W_val + b_val  (16 x 256), then
// per-head G[b][h][:] = mem_feat[b, h*32:(h+1)*32] @ W_out[h*32:(h+1)*32, :]
// ---------------------------------------------------------------------------
__global__ __launch_bounds__(256) void k_build_G(
    const float* __restrict__ memory, const v16h* __restrict__ Fv,
    const float* __restrict__ b_val,  const v16h* __restrict__ Fo,
    float* __restrict__ G /* [16][8][256] */) {
  __shared__ float mf[16][D_MODEL + 4];    // pad -> conflict-free column reads
  const int tid = threadIdx.x, lane = tid & 31, wave = tid >> 5;  // 8 waves

  // Stage 1: 16 N-tiles over 8 waves, K = 256 (8 WMMA steps each).
#pragma unroll
  for (int tt = 0; tt < 2; ++tt) {
    const int t = wave * 2 + tt;
    v8f c = {};
#pragma unroll
    for (int kc = 0; kc < 8; ++kc) {
      v16h a = load_a_f32(memory, D_MODEL, kc * 32, lane);   // memory[0] tile
      v16h b = Fv[(kc * 16 + t) * 32 + lane];                // pre-swizzled f16
      c = wmma_f32_f16(a, b, c);
    }
    const int N  = t * 16 + (lane & 15);
    const int Mb = (lane & 16) ? 8 : 0;
    const float bv = b_val[N];
#pragma unroll
    for (int v = 0; v < 8; ++v) mf[Mb + v][N] = c[v] + bv;
  }
  __syncthreads();

  // Stage 2: 8 heads x 16 N-tiles = 128 single-K-step WMMAs over 8 waves.
#pragma unroll
  for (int j = 0; j < 16; ++j) {
    const int job = j * 8 + wave;          // 0..127
    const int h = job >> 4, t = job & 15;
    v16h a = load_a_f32(&mf[0][0], D_MODEL + 4, h * D_HEAD, lane);
    v16h b = Fo[(h * 16 + t) * 32 + lane];
    v8f c = {};
    c = wmma_f32_f16(a, b, c);
    const int N  = t * 16 + (lane & 15);
    const int Mb = (lane & 16) ? 8 : 0;
#pragma unroll
    for (int v = 0; v < 8; ++v)
      G[((Mb + v) * N_HEADS + h) * D_MODEL + N] = c[v];
  }
}

// ---------------------------------------------------------------------------
// Kernel 2 (grid = NQ): one block per query position n. Rows of one n are a
// contiguous 16x256 f32 tile of `query` ((nq,bs,c) layout).
// logits = q_tile @ W_attn + b_attn -> per-head softmax point-sums s[b][h]
// out[n,b,:] = sum_h s[b][h] * G[b][h][:] + b_out
// ---------------------------------------------------------------------------
__global__ __launch_bounds__(128) void k_attend_out(
    const float* __restrict__ query, const v16h* __restrict__ Fa,
    const float* __restrict__ b_attn, const float* __restrict__ G,
    const float* __restrict__ b_out,  float* __restrict__ out) {
  __shared__ float logits[16][36];         // 16 x 32, padded
  __shared__ float s[16][N_HEADS];
  const int tid = threadIdx.x, lane = tid & 31, wave = tid >> 5;  // 4 waves
  const int n = blockIdx.x;
  const float* qtile = query + (size_t)n * BS * D_MODEL;

  if (wave < 2) {
    // Waves 0/1: one 16-col logit tile each, K = 256; B frags are single
    // 32B/lane loads from the pre-swizzled f16 image (L2-resident).
    const int t = wave;
    v8f c = {};
#pragma unroll
    for (int kc = 0; kc < 8; ++kc) {
      v16h a = load_a_f32(qtile, D_MODEL, kc * 32, lane);
      v16h b = Fa[(kc * 2 + t) * 32 + lane];
      c = wmma_f32_f16(a, b, c);
    }
    const int N  = t * 16 + (lane & 15);
    const int Mb = (lane & 16) ? 8 : 0;
    const float ba = b_attn[N];
#pragma unroll
    for (int v = 0; v < 8; ++v) logits[Mb + v][N] = c[v] + ba;
  } else {
    // Waves 2/3: warm caches with G (used by combine) via global_prefetch.
    const int p = tid - 64;                // 0..63 -> 512 floats each
#pragma unroll
    for (int i = 0; i < 16; ++i)
      __builtin_prefetch(G + p * 512 + i * 32, 0, 0);
  }
  __syncthreads();

  // softmax over each group of 4 points, summed -> s[b][h] (128 jobs)
  {
    const int b = tid >> 3, h = tid & 7;
    const float x0 = logits[b][h * 4 + 0];
    const float x1 = logits[b][h * 4 + 1];
    const float x2 = logits[b][h * 4 + 2];
    const float x3 = logits[b][h * 4 + 3];
    const float m  = fmaxf(fmaxf(x0, x1), fmaxf(x2, x3));
    const float e0 = expf(x0 - m), e1 = expf(x1 - m);
    const float e2 = expf(x2 - m), e3 = expf(x3 - m);
    const float d  = e0 + e1 + e2 + e3;
    s[b][h] = e0 / d + e1 / d + e2 / d + e3 / d;
  }
  __syncthreads();

  // combine + store: out[n,b,c] = sum_h s[b][h]*G[b][h][c] + b_out[c]
  float* orow = out + (size_t)n * BS * D_MODEL;
  for (int idx = tid; idx < BS * D_MODEL; idx += 128) {
    const int b = idx >> 8, cc = idx & 255;
    float acc = b_out[cc];
#pragma unroll
    for (int h = 0; h < N_HEADS; ++h)
      acc += s[b][h] * G[(b * N_HEADS + h) * D_MODEL + cc];
    orow[idx] = acc;
  }
}

extern "C" void kernel_launch(void* const* d_in, const int* in_sizes, int n_in,
                              void* d_out, int out_size, void* d_ws, size_t ws_size,
                              hipStream_t stream) {
  (void)in_sizes; (void)n_in; (void)out_size; (void)ws_size;
  const float* query  = (const float*)d_in[0];
  const float* memory = (const float*)d_in[1];
  // d_in[2] = W_off, d_in[3] = b_off : dead in the reference (offsets deleted)
  const float* W_attn = (const float*)d_in[4];
  const float* b_attn = (const float*)d_in[5];
  const float* W_val  = (const float*)d_in[6];
  const float* b_val  = (const float*)d_in[7];
  const float* W_out  = (const float*)d_in[8];
  const float* b_out  = (const float*)d_in[9];

  char* ws = (char*)d_ws;
  float* G  = (float*)ws;                  // 128 KB
  v16h*  Fa = (v16h*)(ws + WS_FA_OFF);     // 16 KB
  v16h*  Fv = (v16h*)(ws + WS_FV_OFF);     // 128 KB
  v16h*  Fo = (v16h*)(ws + WS_FO_OFF);     // 128 KB

  const int prep_threads = NFRAG_TOTAL * 32;                 // 8704
  k_swizzle_weights<<<(prep_threads + 255) / 256, 256, 0, stream>>>(
      W_attn, W_val, W_out, Fa, Fv, Fo);
  k_build_G<<<1, 256, 0, stream>>>(memory, Fv, b_val, Fo, G);
  k_attend_out<<<NQ, 128, 0, stream>>>(query, Fa, b_attn, G, b_out, (float*)d_out);
}
